// GFCA_9629316678099
// MI455X (gfx1250) — compile-verified, matrix operand order checked by reading, not run on hardware
//
#include <hip/hip_runtime.h>
#include <hip/hip_bf16.h>
#include <math.h>

typedef __attribute__((ext_vector_type(16))) __bf16 v16bf;
typedef __attribute__((ext_vector_type(8)))  float  v8f;

__device__ __forceinline__ __bf16 f2bf(float f) {
  union { float f; unsigned u; } a; a.f = f;
  unsigned r = (a.u + 0x7FFFu + ((a.u >> 16) & 1u)) >> 16;  // RNE truncate
  union { unsigned short s; __bf16 b; } o; o.s = (unsigned short)r;
  return o.b;
}
__device__ __forceinline__ float bfbits2f(unsigned short s) {
  union { unsigned u; float f; } v; v.u = ((unsigned)s) << 16;
  return v.f;
}
// float atomic max via sign-aware integer atomics (m initialized to -inf)
__device__ __forceinline__ void atomicMaxF(float* a, float v) {
  if (v >= 0.0f) atomicMax((int*)a, __float_as_int(v));
  else           atomicMin((unsigned int*)a, __float_as_uint(v));
}
// CDNA5 16-bit A/B fragment K mapping (ISA 7.12.2): pair j (0..7), half = lane>>4
__device__ __forceinline__ int fragKBase(int j, int half) {
  return ((j < 4) ? (2 * j) : (16 + 2 * (j - 4))) + half * 8;
}

// ---------------------------------------------------------------------------
// Kernel 1: fused node projections via WMMA (bf16 in, f32 acc)
//   h = relu(x @ lin_w + lin_b)   (kept in LDS as bf16 A-fragments)
//   Q = h@Wq+bq, K = h@Wk+bk, V = h@Wv+bv, AGG = h@Wskip+bskip
// One wave = one 16-node tile; block = 8 waves = 128 nodes.
// ---------------------------------------------------------------------------
__global__ void __launch_bounds__(256)
node_proj_kernel(const float* __restrict__ x,
                 const float* __restrict__ w_lin, const float* __restrict__ b_lin,
                 const float* __restrict__ w_q,   const float* __restrict__ b_q,
                 const float* __restrict__ w_k,   const float* __restrict__ b_k,
                 const float* __restrict__ w_v,   const float* __restrict__ b_v,
                 const float* __restrict__ w_s,   const float* __restrict__ b_s,
                 float* __restrict__ Q, float* __restrict__ K,
                 float* __restrict__ V, float* __restrict__ AGG, int N)
{
  __shared__ __bf16 sW[5 * 4096];    // transposed: [mat][(c*64)+k]  (c=out, k=in)
  __shared__ float  sB[5 * 64];
  __shared__ __bf16 sH[8 * 16 * 64]; // per-wave h tile, row-major [m][k]

  int tid = threadIdx.x;
  for (int i = tid; i < 4096; i += 256) {
    int k = i >> 6, c = i & 63;
    int d = c * 64 + k;                      // transpose on the fly
    sW[d]            = f2bf(w_lin[i]);
    sW[4096 + d]     = f2bf(w_q[i]);
    sW[2 * 4096 + d] = f2bf(w_k[i]);
    sW[3 * 4096 + d] = f2bf(w_v[i]);
    sW[4 * 4096 + d] = f2bf(w_s[i]);
  }
  if (tid < 64) {
    sB[tid]           = b_lin[tid];
    sB[64 + tid]      = b_q[tid];
    sB[2 * 64 + tid]  = b_k[tid];
    sB[3 * 64 + tid]  = b_v[tid];
    sB[4 * 64 + tid]  = b_s[tid];
  }
  __syncthreads();

  int wave = tid >> 5, lane = tid & 31;
  int half = lane >> 4, m = lane & 15, nn = lane & 15;
  int ntiles = (N + 15) >> 4;
  int tile = blockIdx.x * 8 + wave;
  if (tile >= ntiles) return;
  int row0 = tile * 16;
  int arow = min(row0 + m, N - 1);

  // A-fragments of the x tile (K=0..31 and K=32..63)
  v16bf a_lo, a_hi;
#pragma unroll
  for (int j = 0; j < 8; ++j) {
    int kb = fragKBase(j, half);
    const float* p = x + (size_t)arow * 64 + kb;
    a_lo[2 * j] = f2bf(p[0]);  a_lo[2 * j + 1] = f2bf(p[1]);
    a_hi[2 * j] = f2bf(p[32]); a_hi[2 * j + 1] = f2bf(p[33]);
  }

  __bf16* hT = sH + wave * (16 * 64);

  // h = relu(x @ lin_w + b) -> LDS (bf16)
#pragma unroll
  for (int cg = 0; cg < 4; ++cg) {
    float bias = sB[cg * 16 + nn];
    v8f acc;
#pragma unroll
    for (int i = 0; i < 8; ++i) acc[i] = bias;
    v16bf b0, b1;
    const __bf16* wc = sW + (cg * 16 + nn) * 64;
#pragma unroll
    for (int j = 0; j < 8; ++j) {
      int kb = fragKBase(j, half);
      b0[2 * j] = wc[kb];      b0[2 * j + 1] = wc[kb + 1];
      b1[2 * j] = wc[kb + 32]; b1[2 * j + 1] = wc[kb + 33];
    }
    acc = __builtin_amdgcn_wmma_f32_16x16x32_bf16(false, a_lo, false, b0, (short)0, acc, false, false);
    acc = __builtin_amdgcn_wmma_f32_16x16x32_bf16(false, a_hi, false, b1, (short)0, acc, false, false);
#pragma unroll
    for (int i = 0; i < 8; ++i) {
      int mr = half * 8 + i;                       // D layout: row = i + 8*half
      hT[mr * 64 + cg * 16 + nn] = f2bf(fmaxf(acc[i], 0.0f));
    }
  }

  // reload h tile as A-fragments (same-wave LDS region, compiler inserts s_wait_dscnt)
  v16bf h_lo, h_hi;
#pragma unroll
  for (int j = 0; j < 8; ++j) {
    int kb = fragKBase(j, half);
    h_lo[2 * j] = hT[m * 64 + kb];      h_lo[2 * j + 1] = hT[m * 64 + kb + 1];
    h_hi[2 * j] = hT[m * 64 + kb + 32]; h_hi[2 * j + 1] = hT[m * 64 + kb + 33];
  }

  float* dsts[4] = {Q, K, V, AGG};
#pragma unroll
  for (int mat = 1; mat <= 4; ++mat) {
    float* dst = dsts[mat - 1];
#pragma unroll
    for (int cg = 0; cg < 4; ++cg) {
      float bias = sB[mat * 64 + cg * 16 + nn];
      v8f acc;
#pragma unroll
      for (int i = 0; i < 8; ++i) acc[i] = bias;
      v16bf b0, b1;
      const __bf16* wc = sW + mat * 4096 + (cg * 16 + nn) * 64;
#pragma unroll
      for (int j = 0; j < 8; ++j) {
        int kb = fragKBase(j, half);
        b0[2 * j] = wc[kb];      b0[2 * j + 1] = wc[kb + 1];
        b1[2 * j] = wc[kb + 32]; b1[2 * j + 1] = wc[kb + 33];
      }
      acc = __builtin_amdgcn_wmma_f32_16x16x32_bf16(false, h_lo, false, b0, (short)0, acc, false, false);
      acc = __builtin_amdgcn_wmma_f32_16x16x32_bf16(false, h_hi, false, b1, (short)0, acc, false, false);
#pragma unroll
      for (int i = 0; i < 8; ++i) {
        int r = row0 + half * 8 + i;
        if (r < N) dst[(size_t)r * 64 + cg * 16 + nn] = acc[i];
      }
    }
  }
}

// ---------------------------------------------------------------------------
// Kernel 2: per-edge attention logits + scatter max. One wave32 per edge.
// lane = channel c and c+32 (head0 / head1).
// ---------------------------------------------------------------------------
__global__ void __launch_bounds__(256)
edge_alpha_kernel(const int* __restrict__ ei, const float* __restrict__ node_time,
                  const float* __restrict__ edge_time,
                  const float* __restrict__ time_w, const float* __restrict__ time_b,
                  const float* __restrict__ We, const float* __restrict__ be,
                  const float* __restrict__ Q, const float* __restrict__ K,
                  float* __restrict__ alpha, float* __restrict__ mmax, int E)
{
  __shared__ float sWe[32 * 64];
  __shared__ float stw[32], stb[32], sbe[64];
  int tid = threadIdx.x;
  for (int i = tid; i < 2048; i += 256) sWe[i] = We[i];
  if (tid < 32) { stw[tid] = time_w[tid]; stb[tid] = time_b[tid]; }
  if (tid < 64) sbe[tid] = be[tid];
  __syncthreads();

  int wave = tid >> 5, lane = tid & 31;
  int e = blockIdx.x * 8 + wave;
  if (e >= E) return;
  int src = ei[e], dst = ei[E + e];
  float rel = node_time[src] - edge_time[e];
  float enc = __cosf(rel * stw[lane] + stb[lane]);
  float acc0 = sbe[lane], acc1 = sbe[lane + 32];
  for (int j = 0; j < 32; ++j) {
    float ej = __shfl(enc, j, 32);
    acc0 += ej * sWe[j * 64 + lane];
    acc1 += ej * sWe[j * 64 + 32 + lane];
  }
  size_t so = (size_t)src * 64, dof = (size_t)dst * 64;
  float p0 = Q[dof + lane]      * (K[so + lane]      + acc0);
  float p1 = Q[dof + 32 + lane] * (K[so + 32 + lane] + acc1);
#pragma unroll
  for (int off = 16; off > 0; off >>= 1) {
    p0 += __shfl_xor(p0, off, 32);
    p1 += __shfl_xor(p1, off, 32);
  }
  if (lane == 0) {
    const float sc = 0.17677669529663687f;   // 1/sqrt(32)
    p0 *= sc; p1 *= sc;
    alpha[2 * (size_t)e]     = p0;
    alpha[2 * (size_t)e + 1] = p1;
    atomicMaxF(&mmax[2 * dst],     p0);
    atomicMaxF(&mmax[2 * dst + 1], p1);
  }
}

// ---------------------------------------------------------------------------
// Kernel 3: ea = exp(alpha - m[dst]); denom[dst] += ea.  One thread per (edge,head).
// ---------------------------------------------------------------------------
__global__ void edge_softmax_kernel(const int* __restrict__ ei, float* __restrict__ alpha,
                                    const float* __restrict__ mmax, float* __restrict__ denom, int E)
{
  int i = blockIdx.x * blockDim.x + threadIdx.x;
  if (i >= 2 * E) return;
  int e = i >> 1, h = i & 1;
  int dst = ei[E + e];
  float ea = __expf(alpha[i] - mmax[2 * dst + h]);
  alpha[i] = ea;
  atomicAdd(&denom[2 * dst + h], ea);
}

// ---------------------------------------------------------------------------
// Kernel 4: weighted V scatter-add (recomputes the edge projection from LDS weights).
// ---------------------------------------------------------------------------
__global__ void __launch_bounds__(256)
edge_agg_kernel(const int* __restrict__ ei, const float* __restrict__ node_time,
                const float* __restrict__ edge_time,
                const float* __restrict__ time_w, const float* __restrict__ time_b,
                const float* __restrict__ We, const float* __restrict__ be,
                const float* __restrict__ alpha, const float* __restrict__ denom,
                const float* __restrict__ V, float* __restrict__ AGG, int E)
{
  __shared__ float sWe[32 * 64];
  __shared__ float stw[32], stb[32], sbe[64];
  int tid = threadIdx.x;
  for (int i = tid; i < 2048; i += 256) sWe[i] = We[i];
  if (tid < 32) { stw[tid] = time_w[tid]; stb[tid] = time_b[tid]; }
  if (tid < 64) sbe[tid] = be[tid];
  __syncthreads();

  int wave = tid >> 5, lane = tid & 31;
  int e = blockIdx.x * 8 + wave;
  if (e >= E) return;
  int src = ei[e], dst = ei[E + e];
  float a0 = alpha[2 * (size_t)e]     / (denom[2 * dst]     + 1e-16f);
  float a1 = alpha[2 * (size_t)e + 1] / (denom[2 * dst + 1] + 1e-16f);
  float rel = node_time[src] - edge_time[e];
  float enc = __cosf(rel * stw[lane] + stb[lane]);
  float acc0 = sbe[lane], acc1 = sbe[lane + 32];
  for (int j = 0; j < 32; ++j) {
    float ej = __shfl(enc, j, 32);
    acc0 += ej * sWe[j * 64 + lane];
    acc1 += ej * sWe[j * 64 + 32 + lane];
  }
  size_t so = (size_t)src * 64, dof = (size_t)dst * 64;
  float v0 = V[so + lane]      + acc0;
  float v1 = V[so + 32 + lane] + acc1;
  atomicAdd(&AGG[dof + lane],      a0 * v0);
  atomicAdd(&AGG[dof + 32 + lane], a1 * v1);
}

// ---------------------------------------------------------------------------
// Kernel 5: context attention + combine + log_softmax. One thread per node,
// all weights staged in LDS (uniform c => broadcast LDS reads, no conflicts).
// ---------------------------------------------------------------------------
__global__ void __launch_bounds__(256)
node_post_kernel(const float* __restrict__ AGG,
                 const float* __restrict__ nmoti, const float* __restrict__ ndeg,
                 const int* __restrict__ y, const unsigned char* __restrict__ mask,
                 const float* __restrict__ tf_w, const float* __restrict__ tf_b,
                 const float* __restrict__ deg_w, const float* __restrict__ deg_b,
                 const float* __restrict__ wenc_w, const float* __restrict__ wenc_b,
                 const float* __restrict__ wx_w, const float* __restrict__ wx_b,
                 const float* __restrict__ comb_w, const float* __restrict__ comb_b,
                 const float* __restrict__ out_w, const float* __restrict__ out_b,
                 float* __restrict__ out, int N)
{
  __shared__ float sWenc[4096];
  __shared__ float sWx[4096];
  __shared__ unsigned short sComb[8192];     // bf16 bits, saves LDS
  __shared__ float sOutW[128];
  __shared__ float sTfw[64], sTfb[64], sDegw[64], sDegb[64];
  __shared__ float sWencB[64], sWxB[64], sCombB[64], sOutB[2];

  int tid = threadIdx.x;
  for (int i = tid; i < 4096; i += 256) { sWenc[i] = wenc_w[i]; sWx[i] = wx_w[i]; }
  for (int i = tid; i < 8192; i += 256) {
    union { float f; unsigned u; } a; a.f = comb_w[i];
    sComb[i] = (unsigned short)((a.u + 0x7FFFu + ((a.u >> 16) & 1u)) >> 16);
  }
  if (tid < 128) sOutW[tid] = out_w[tid];
  if (tid < 64) {
    sTfw[tid] = tf_w[tid]; sTfb[tid] = tf_b[tid];
    sDegw[tid] = deg_w[tid]; sDegb[tid] = deg_b[tid];
    sWencB[tid] = wenc_b[tid]; sWxB[tid] = wx_b[tid]; sCombB[tid] = comb_b[tid];
  }
  if (tid < 2) sOutB[tid] = out_b[tid];
  __syncthreads();

  int n = blockIdx.x * 256 + tid;
  if (n >= N) return;

  float h1[64];
#pragma unroll
  for (int c = 0; c < 64; ++c) h1[c] = AGG[(size_t)n * 64 + c];
  float t = nmoti[n], dg = ndeg[n];

  // size-2 context attention scores
  float s0 = 0.f, s1 = 0.f;
  for (int c = 0; c < 64; ++c) {
    float ax = sWxB[c], a0 = sWencB[c], a1 = sWencB[c];
#pragma unroll
    for (int j = 0; j < 64; ++j) {
      float we = sWenc[j * 64 + c];
      ax += h1[j] * sWx[j * 64 + c];
      a0 += (t  * sTfw[j]  + sTfb[j])  * we;
      a1 += (dg * sDegw[j] + sDegb[j]) * we;
    }
    float xp = tanhf(ax);
    s0 += tanhf(a0) * xp;
    s1 += tanhf(a1) * xp;
  }
  float mx = fmaxf(s0, s1);
  float w0 = __expf(s0 - mx), w1 = __expf(s1 - mx);
  float inv = 1.0f / (w0 + w1);
  float p0 = w0 * inv, p1 = w1 * inv;

  // combine + output logits
  float l0 = sOutB[0], l1 = sOutB[1];
  for (int c = 0; c < 64; ++c) {
    float acc = sCombB[c];
#pragma unroll
    for (int j = 0; j < 64; ++j)
      acc += h1[j] * bfbits2f(sComb[j * 64 + c]);
#pragma unroll
    for (int j = 0; j < 64; ++j) {
      float ctx = (t * sTfw[j] + sTfb[j]) * p0 + (dg * sDegw[j] + sDegb[j]) * p1;
      acc += ctx * bfbits2f(sComb[(64 + j) * 64 + c]);
    }
    out[(size_t)n * 64 + c] = acc;
    l0 += acc * sOutW[c * 2 + 0];
    l1 += acc * sOutW[c * 2 + 1];
  }
  float lm = fmaxf(l0, l1);
  float lse = lm + __logf(__expf(l0 - lm) + __expf(l1 - lm));
  size_t base = (size_t)N * 64;
  out[base + 2 * (size_t)n]     = l0 - lse;
  out[base + 2 * (size_t)n + 1] = l1 - lse;
  out[base + 2 * (size_t)N + n] = (float)y[n];
  out[base + 3 * (size_t)N + n] = mask[n] ? 1.0f : 0.0f;
}

__global__ void init_md_kernel(float* __restrict__ mmax, float* __restrict__ denom, int n2)
{
  int i = blockIdx.x * blockDim.x + threadIdx.x;
  if (i < n2) { mmax[i] = -__builtin_huge_valf(); denom[i] = 0.0f; }
}

// ---------------------------------------------------------------------------
extern "C" void kernel_launch(void* const* d_in, const int* in_sizes, int n_in,
                              void* d_out, int out_size, void* d_ws, size_t ws_size,
                              hipStream_t stream)
{
  const float* x         = (const float*)d_in[0];
  const int*   ei        = (const int*)d_in[1];
  const float* node_time = (const float*)d_in[2];
  const float* edge_time = (const float*)d_in[3];
  const float* nmoti     = (const float*)d_in[4];
  const float* ndeg      = (const float*)d_in[5];
  const int*   y         = (const int*)d_in[6];
  const unsigned char* mask = (const unsigned char*)d_in[7];
  const float* time_w = (const float*)d_in[8];
  const float* time_b = (const float*)d_in[9];
  const float* tf_w   = (const float*)d_in[10];
  const float* tf_b   = (const float*)d_in[11];
  const float* deg_w  = (const float*)d_in[12];
  const float* deg_b  = (const float*)d_in[13];
  const float* lin_w  = (const float*)d_in[14];
  const float* lin_b  = (const float*)d_in[15];
  const float* Wq = (const float*)d_in[16];  const float* bq = (const float*)d_in[17];
  const float* Wk = (const float*)d_in[18];  const float* bk = (const float*)d_in[19];
  const float* Wv = (const float*)d_in[20];  const float* bv = (const float*)d_in[21];
  const float* We = (const float*)d_in[22];  const float* be = (const float*)d_in[23];
  const float* Wskip = (const float*)d_in[24]; const float* bskip = (const float*)d_in[25];
  const float* wenc_w = (const float*)d_in[26]; const float* wenc_b = (const float*)d_in[27];
  const float* wx_w   = (const float*)d_in[28]; const float* wx_b   = (const float*)d_in[29];
  const float* comb_w = (const float*)d_in[30]; const float* comb_b = (const float*)d_in[31];
  const float* out_w  = (const float*)d_in[32]; const float* out_b  = (const float*)d_in[33];

  const int N = in_sizes[2];   // node_time count
  const int E = in_sizes[3];   // edge_time count

  // workspace layout (~112 MB for N=1e5, E=1e6)
  float* Q     = (float*)d_ws;
  float* K     = Q + (size_t)N * 64;
  float* V     = K + (size_t)N * 64;
  float* AGG   = V + (size_t)N * 64;     // initialized with skip projection by kernel 1
  float* mmax  = AGG + (size_t)N * 64;   // [N,2]
  float* denom = mmax + (size_t)2 * N;   // [N,2]
  float* alpha = denom + (size_t)2 * N;  // [E,2]

  init_md_kernel<<<(2 * N + 255) / 256, 256, 0, stream>>>(mmax, denom, 2 * N);

  int ntiles = (N + 15) / 16;
  node_proj_kernel<<<(ntiles + 7) / 8, 256, 0, stream>>>(
      x, lin_w, lin_b, Wq, bq, Wk, bk, Wv, bv, Wskip, bskip, Q, K, V, AGG, N);

  edge_alpha_kernel<<<(E + 7) / 8, 256, 0, stream>>>(
      ei, node_time, edge_time, time_w, time_b, We, be, Q, K, alpha, mmax, E);

  edge_softmax_kernel<<<(2 * E + 255) / 256, 256, 0, stream>>>(ei, alpha, mmax, denom, E);

  edge_agg_kernel<<<(E + 7) / 8, 256, 0, stream>>>(
      ei, node_time, edge_time, time_w, time_b, We, be, alpha, denom, V, AGG, E);

  node_post_kernel<<<(N + 255) / 256, 256, 0, stream>>>(
      AGG, nmoti, ndeg, y, mask, tf_w, tf_b, deg_w, deg_b,
      wenc_w, wenc_b, wx_w, wx_b, comb_w, comb_b, out_w, out_b,
      (float*)d_out, N);
}
